// SparseViT_83906481095482
// MI455X (gfx1250) — matrix operand order repaired and checked by reading, not compiled, not run on hardware
//
#include <hip/hip_runtime.h>

// ---- model constants ----
#define IMGSZ 24
#define BATCH 2
#define CHN   3
#define NTOK  577          // IMG*IMG + 1
#define EMB   512
#define NDEPTH 2
#define NH    8
#define DHD   64
#define MLPD  2048
#define NCLS  1000
#define SWIN  32
#define CBK   16
#define CSS   4
#define SBL   64
#define KSEL4 4
#define NCMP  145          // ceil(577/4)
#define NCMPP 160          // NCMP padded to mult-32 (zero-filled)
#define NSB   10           // ceil(577/64)
#define NEG_BIG (-3.0e38f)

typedef __attribute__((ext_vector_type(16))) _Float16 v16h;
typedef __attribute__((ext_vector_type(8)))  _Float16 v8h;
typedef __attribute__((ext_vector_type(8)))  float    v8f;
typedef __attribute__((ext_vector_type(4)))  float    v4f;

enum { F_BIAS = 1, F_GELU = 2, F_SIG = 4, F_RES = 8 };

__device__ __forceinline__ int iminc(int a, int b) { return a < b ? a : b; }

// =====================================================================
// Pack B (KxN f32, arbitrary strides incl. transpose) into f16
// column-major tiles: Bp[n*Kp + k], n padded to Np64 (mult 64, = grid
// strip coverage), k padded to Kp (mult 32), pads zero-filled. After
// this, GEMM B-fragment loads are two aligned b128 per 16x16 tile.
// =====================================================================
__global__ void pack_b(const float* __restrict__ B, _Float16* __restrict__ Bp,
                       int K, int N, int Kp, int Np64,
                       int ldbK, int ldbN, int innerB,
                       long long sBb, long long sBh, long long sBp)
{
    int z  = blockIdx.z;
    int zb = z / innerB;
    int zh = z - zb * innerB;
    B  += zb * sBb + zh * sBh;
    Bp += (long long)z * sBp;
    int total = Np64 * Kp;
    int idx = blockIdx.x * blockDim.x + threadIdx.x;
    if (idx >= total) return;
    int k = idx % Kp;
    int n = idx / Kp;
    float v = (k < K && n < N) ? B[(size_t)k * ldbK + (size_t)n * ldbN] : 0.f;
    Bp[idx] = (_Float16)v;
}

// Per-k-step fragment set: raw f32 A (4 x b128) + 4 f16 B fragments.
struct KFrags {
    v4f  ar[4];
    v16h bf[4];
};

__device__ __forceinline__ v16h load_bfrag(const _Float16* bp)
{
    v8h b0 = *(const v8h*)(bp);
    v8h b1 = *(const v8h*)(bp + 16);
    v16h bf;
#pragma unroll
    for (int e = 0; e < 8; ++e) { bf[e] = b0[e]; bf[e + 8] = b1[e]; }
    return bf;
}

__device__ __forceinline__ void load_frags(KFrags& f, const float* Abase,
                                           const _Float16* const* Bl, int kk)
{
    const float* pn = Abase + kk;
    f.ar[0] = *(const v4f*)(pn);
    f.ar[1] = *(const v4f*)(pn + 4);
    f.ar[2] = *(const v4f*)(pn + 16);
    f.ar[3] = *(const v4f*)(pn + 20);
#pragma unroll
    for (int s = 0; s < 4; ++s) f.bf[s] = load_bfrag(Bl[s] + kk);
}

__device__ __forceinline__ void compute_frags(const KFrags& f, v8f acc[4])
{
    v16h af;
#pragma unroll
    for (int e = 0; e < 4; ++e) {
        af[e]      = (_Float16)f.ar[0][e];
        af[e + 4]  = (_Float16)f.ar[1][e];
        af[e + 8]  = (_Float16)f.ar[2][e];
        af[e + 12] = (_Float16)f.ar[3][e];
    }
#pragma unroll
    for (int s = 0; s < 4; ++s)
        acc[s] = __builtin_amdgcn_wmma_f32_16x16x32_f16(
            false, af, false, f.bf[s], (short)0, acc[s], false, false);
}

// =====================================================================
// Batched/strided WMMA GEMM: C = epilogue(alpha * A(MxKp f32) * Bp).
// One wave computes a 16x64 C strip (4 x 16x16 tiles). The K loop is
// manually unrolled x2 into a register ping-pong (f0/f1): while one
// fragment set feeds 4 back-to-back WMMAs, the other set's 12 b128
// loads are in flight -- no rotation copies, partial loadcnt waits.
// Kp is a multiple of 32 (A buffers padded/zero where needed): zero
// guards in the loop; only C stores are bounds-checked.
// Fragment layouts per CDNA5 ISA 7.12.2 (wave32):
//   A (16x32 f16): lane L: M=L&15 ; elem e -> K = e + 8*((e>>3)+(L>>4))
//   B (32x16 f16): lane L: N=L&15 ; elem e -> K = e + 8*((e>>3)+(L>>4))
//   C (16x16 f32): lane L: N=L&15 ; elem r -> M = r + 8*(L>>4)
// =====================================================================
__global__ __launch_bounds__(32)
void wmma_gemm(const float* __restrict__ A, const _Float16* __restrict__ Bp,
               const float* __restrict__ bias, const float* __restrict__ res,
               float* __restrict__ C,
               int M, int N, int Kp, int lda, int ldc,
               int innerB,
               long long sAb, long long sAh, long long sBp,
               long long sCb, long long sCh,
               float alpha, int flags)
{
    int z  = blockIdx.z;
    int zb = z / innerB;
    int zh = z - zb * innerB;
    A  += zb * sAb + zh * sAh;
    Bp += (long long)z * sBp;
    C  += zb * sCb + zh * sCh;
    if (res) res += zb * sCb + zh * sCh;

    const int lane = threadIdx.x;        // 0..31
    const int hi   = lane >> 4;          // half-wave select
    const int l15  = lane & 15;
    const int tm   = blockIdx.y * 16;
    const int tn0  = blockIdx.x * 64;

    // clamped A row (rows beyond M are computed but never stored)
    const float* Abase =
        A + (size_t)iminc(tm + l15, M - 1) * lda + 8 * hi;
    const _Float16* Bl[4];
#pragma unroll
    for (int s = 0; s < 4; ++s)
        Bl[s] = Bp + (size_t)(tn0 + s * 16 + l15) * Kp + 8 * hi;

    v8f acc[4];
#pragma unroll
    for (int s = 0; s < 4; ++s)
        acc[s] = (v8f){0.f, 0.f, 0.f, 0.f, 0.f, 0.f, 0.f, 0.f};

    // ---- ping-pong pipeline over K (all branches wave-uniform) ----
    KFrags f0, f1;
    load_frags(f0, Abase, Bl, 0);
    int kk = 0;
    for (; kk + 64 <= Kp; kk += 64) {
        load_frags(f1, Abase, Bl, kk + 32);
        compute_frags(f0, acc);
        if (kk + 64 < Kp) load_frags(f0, Abase, Bl, kk + 64);
        compute_frags(f1, acc);
    }
    if (kk < Kp) compute_frags(f0, acc);   // odd-step tail (e.g. Kp=160)

#pragma unroll
    for (int s = 0; s < 4; ++s) {
        int ncol = tn0 + s * 16 + l15;
        if (ncol >= N) continue;
#pragma unroll
        for (int r = 0; r < 8; ++r) {
            int m = tm + r + 8 * hi;
            if (m < M) {
                float vv = acc[s][r] * alpha;
                if (flags & F_BIAS) vv += bias[ncol];
                if (flags & F_GELU) {
                    float x3 = vv * vv * vv;
                    vv = 0.5f * vv *
                         (1.f + tanhf(0.7978845608028654f *
                                      (vv + 0.044715f * x3)));
                }
                if (flags & F_SIG) vv = 1.f / (1.f + __expf(-vv));
                if (flags & F_RES) vv += res[(size_t)m * ldc + ncol];
                C[(size_t)m * ldc + ncol] = vv;
            }
        }
    }
}

// =====================================================================
// Patch embedding: x[b,n,:] = (n==0 ? cls : pix(b,n-1) @ Wpatch + bpatch) + pos[n]
// =====================================================================
__global__ void patch_embed(const float* __restrict__ img,
                            const float* __restrict__ pw,
                            const float* __restrict__ pb,
                            const float* __restrict__ pos,
                            const float* __restrict__ cls,
                            float* __restrict__ x)
{
    int row = blockIdx.x;               // b*NTOK + n
    int b = row / NTOK, n = row % NTOK;
    for (int d = threadIdx.x; d < EMB; d += blockDim.x) {
        float v;
        if (n == 0) {
            v = cls[d];
        } else {
            int p = n - 1;
            v = img[(b * CHN + 0) * (IMGSZ * IMGSZ) + p] * pw[0 * EMB + d]
              + img[(b * CHN + 1) * (IMGSZ * IMGSZ) + p] * pw[1 * EMB + d]
              + img[(b * CHN + 2) * (IMGSZ * IMGSZ) + p] * pw[2 * EMB + d]
              + pb[d];
        }
        x[(size_t)row * EMB + d] = v + pos[(size_t)n * EMB + d];
    }
}

// =====================================================================
// LayerNorm over EMB, arbitrary row strides (tokens rows + cls head rows)
// =====================================================================
__global__ void ln_kernel(const float* __restrict__ X,
                          const float* __restrict__ gam,
                          const float* __restrict__ bet,
                          float* __restrict__ Y,
                          long long inStride, long long outStride)
{
    __shared__ float red[256];
    int row = blockIdx.x, t = threadIdx.x;
    const float* xr = X + (size_t)row * inStride;
    float* yr = Y + (size_t)row * outStride;

    float s = 0.f;
    for (int j = t; j < EMB; j += 256) s += xr[j];
    red[t] = s; __syncthreads();
    for (int st = 128; st > 0; st >>= 1) {
        if (t < st) red[t] += red[t + st];
        __syncthreads();
    }
    float mean = red[0] / (float)EMB; __syncthreads();

    float vs = 0.f;
    for (int j = t; j < EMB; j += 256) { float d = xr[j] - mean; vs += d * d; }
    red[t] = vs; __syncthreads();
    for (int st = 128; st > 0; st >>= 1) {
        if (t < st) red[t] += red[t + st];
        __syncthreads();
    }
    float inv = rsqrtf(red[0] / (float)EMB + 1e-5f);

    for (int j = t; j < EMB; j += 256)
        yr[j] = (xr[j] - mean) * inv * gam[j] + bet[j];
}

// =====================================================================
// Build compressed windows: kw[b,h,i,j,:] = kpad[b,h,i*4+j,:] + kpe[j,:]
// Flat layout (b*NH, NCMP, CBK, DHD)
// =====================================================================
__global__ void build_kw_vw(const float* __restrict__ kbuf,
                            const float* __restrict__ vbuf,
                            const float* __restrict__ kpe,
                            const float* __restrict__ vpe,
                            float* __restrict__ kw, float* __restrict__ vw)
{
    size_t idx = (size_t)blockIdx.x * blockDim.x + threadIdx.x;
    const size_t total = (size_t)BATCH * NH * NCMP * CBK * DHD;
    if (idx >= total) return;
    int d = idx & (DHD - 1);
    size_t r = idx >> 6;
    int j = r & (CBK - 1); r >>= 4;
    int ic = (int)(r % NCMP);
    int z  = (int)(r / NCMP);           // b*NH + h
    int b = z >> 3, h = z & 7;
    int tok = ic * CSS + j;
    float kv = 0.f, vv = 0.f;
    if (tok < NTOK) {
        size_t base = ((size_t)(b * NTOK + tok)) * EMB + h * DHD + d;
        kv = kbuf[base]; vv = vbuf[base];
    }
    kw[idx] = kv + kpe[j * DHD + d];
    vw[idx] = vv + vpe[j * DHD + d];
}

// ====== row softmax over nc cols, stored with ncPad stride (pad -> 0) ======
__global__ void softmax_rows(float* __restrict__ S, int nc, int ncPad)
{
    __shared__ float red[256];
    int row = blockIdx.x, t = threadIdx.x;
    float* s = S + (size_t)row * ncPad;
    float v = (t < nc) ? s[t] : NEG_BIG;
    red[t] = v; __syncthreads();
    for (int st = 128; st > 0; st >>= 1) {
        if (t < st) red[t] = fmaxf(red[t], red[t + st]);
        __syncthreads();
    }
    float mx = red[0]; __syncthreads();
    float e = (t < nc) ? __expf(v - mx) : 0.f;
    red[t] = e; __syncthreads();
    for (int st = 128; st > 0; st >>= 1) {
        if (t < st) red[t] += red[t + st];
        __syncthreads();
    }
    if (t < ncPad) s[t] = (t < nc) ? e / red[0] : 0.f;
}

// ======= bucket importance over NSB blocks + top-KSEL selection ========
__global__ void topk_kernel(const float* __restrict__ pc,
                            int* __restrict__ selb, int nrows)
{
    int row = blockIdx.x * blockDim.x + threadIdx.x;
    if (row >= nrows) return;
    const float* p = pc + (size_t)row * NCMPP;
    float imp[NSB];
#pragma unroll
    for (int s = 0; s < NSB; ++s) imp[s] = 0.f;
    for (int j = 0; j < NCMP; ++j) {
        int c = (j * CSS + CBK / 2) >> 6;   // (j*4+8)/64
        if (c > NSB - 1) c = NSB - 1;
        imp[c] += p[j];
    }
    bool used[NSB] = {};
#pragma unroll
    for (int t = 0; t < KSEL4; ++t) {
        int best = 0; float bv = NEG_BIG;
        for (int s = 0; s < NSB; ++s)
            if (!used[s] && imp[s] > bv) { bv = imp[s]; best = s; }
        used[best] = true;
        selb[(size_t)row * KSEL4 + t] = best;
    }
}

// =====================================================================
// Fused selected-block + sliding-window attention + gating + merge.
// One block per (b,h,i); 256 threads. Selected: 4 blocks x 64 tokens.
// =====================================================================
__global__ void attn_combine(const float* __restrict__ qb,
                             const float* __restrict__ kb,
                             const float* __restrict__ vb,
                             const float* __restrict__ outc,
                             const float* __restrict__ gb,
                             const int*   __restrict__ selb,
                             float* __restrict__ merged)
{
    __shared__ float qs[DHD];
    __shared__ float sc_s[256];
    __shared__ float sw_s[SWIN];
    __shared__ float red[256];

    int row = blockIdx.x;               // (b*NH + h)*NTOK + i
    int i  = row % NTOK;
    int bh = row / NTOK;
    int h = bh & 7, b = bh >> 3;
    int t = threadIdx.x;

    const float* qrow = qb + ((size_t)(b * NTOK + i)) * EMB + h * DHD;
    if (t < DHD) qs[t] = qrow[t];
    __syncthreads();

    // ---- selected-block scores ----
    int blk = t >> 6, off = t & 63;
    int sel = selb[(size_t)row * KSEL4 + blk];
    int tok = sel * SBL + off;
    bool valid = tok < NTOK;
    int tokc = valid ? tok : NTOK - 1;
    {
        const float* krow = kb + ((size_t)(b * NTOK + tokc)) * EMB + h * DHD;
        float s = 0.f;
#pragma unroll 16
        for (int d = 0; d < DHD; ++d) s += qs[d] * krow[d];
        sc_s[t] = valid ? s * 0.125f : NEG_BIG;
    }
    // ---- window scores ----
    if (t < SWIN) {
        int pos = i - SWIN / 2 + t;
        bool wm = (pos >= 0) && (pos < NTOK);
        int pc2 = pos < 0 ? 0 : (pos > NTOK - 1 ? NTOK - 1 : pos);
        const float* krow = kb + ((size_t)(b * NTOK + pc2)) * EMB + h * DHD;
        float s = 0.f;
#pragma unroll 16
        for (int d = 0; d < DHD; ++d) s += qs[d] * krow[d];
        sw_s[t] = wm ? s * 0.125f : NEG_BIG;
    }
    __syncthreads();

    // ---- softmax over 256 selected scores ----
    red[t] = sc_s[t]; __syncthreads();
    for (int st = 128; st > 0; st >>= 1) {
        if (t < st) red[t] = fmaxf(red[t], red[t + st]);
        __syncthreads();
    }
    float mx = red[0]; __syncthreads();
    float e = __expf(sc_s[t] - mx);
    red[t] = e; __syncthreads();
    for (int st = 128; st > 0; st >>= 1) {
        if (t < st) red[t] += red[t + st];
        __syncthreads();
    }
    float inv = 1.f / red[0]; __syncthreads();
    sc_s[t] = e * inv;

    // ---- softmax over 32 window scores (serial, tiny) ----
    if (t == 0) {
        float wm = NEG_BIG;
        for (int j = 0; j < SWIN; ++j) wm = fmaxf(wm, sw_s[j]);
        float ws = 0.f;
        for (int j = 0; j < SWIN; ++j) { sw_s[j] = __expf(sw_s[j] - wm); ws += sw_s[j]; }
        float wi = 1.f / ws;
        for (int j = 0; j < SWIN; ++j) sw_s[j] *= wi;
    }
    __syncthreads();

    // ---- out_s partials: thread t -> dim (t&63), token group (t>>6) ----
    {
        int d = t & 63, g = t >> 6;
        int sg = selb[(size_t)row * KSEL4 + g];
        float part = 0.f;
        for (int j = 0; j < SBL; ++j) {
            int tk = sg * SBL + j;
            int tc = tk < NTOK ? tk : NTOK - 1;
            part += sc_s[g * SBL + j] *
                    vb[((size_t)(b * NTOK + tc)) * EMB + h * DHD + d];
        }
        red[t] = part;
    }
    __syncthreads();

    if (t < DHD) {
        float os = red[t] + red[t + 64] + red[t + 128] + red[t + 192];
        float ow = 0.f;
        for (int j = 0; j < SWIN; ++j) {
            int pos = i - SWIN / 2 + j;
            int pc2 = pos < 0 ? 0 : (pos > NTOK - 1 ? NTOK - 1 : pos);
            ow += sw_s[j] * vb[((size_t)(b * NTOK + pc2)) * EMB + h * DHD + t];
        }
        const float* gr = gb + ((size_t)(b * NTOK + i)) * (3 * NH) + h * 3;
        float oc = outc[((size_t)bh * NTOK + i) * DHD + t];
        merged[((size_t)(b * NTOK + i)) * EMB + h * DHD + t] =
            gr[0] * oc + gr[1] * os + gr[2] * ow;
    }
}

// =====================================================================
extern "C" void kernel_launch(void* const* d_in, const int* in_sizes, int n_in,
                              void* d_out, int out_size, void* d_ws, size_t ws_size,
                              hipStream_t stream)
{
    (void)in_sizes; (void)n_in; (void)out_size; (void)ws_size;
    const float* img     = (const float*)d_in[0];
    const float* patch_w = (const float*)d_in[1];
    const float* patch_b = (const float*)d_in[2];
    const float* pos_emb = (const float*)d_in[3];
    const float* cls_tok = (const float*)d_in[4];
    const float* ln1_g   = (const float*)d_in[5];
    const float* ln1_b   = (const float*)d_in[6];
    const float* Wq      = (const float*)d_in[7];
    const float* Wk      = (const float*)d_in[8];
    const float* Wv      = (const float*)d_in[9];
    const float* Wg      = (const float*)d_in[10];
    const float* Wo      = (const float*)d_in[11];
    const float* kpe     = (const float*)d_in[12];
    const float* vpe     = (const float*)d_in[13];
    const float* Wkc     = (const float*)d_in[14];
    const float* Wvc     = (const float*)d_in[15];
    const float* ln2_g   = (const float*)d_in[16];
    const float* ln2_b   = (const float*)d_in[17];
    const float* ff_w1   = (const float*)d_in[18];
    const float* ff_b1   = (const float*)d_in[19];
    const float* ff_w2   = (const float*)d_in[20];
    const float* ff_b2   = (const float*)d_in[21];
    const float* hln_g   = (const float*)d_in[22];
    const float* hln_b   = (const float*)d_in[23];
    const float* head_w  = (const float*)d_in[24];
    const float* head_b  = (const float*)d_in[25];
    float* out = (float*)d_out;

    const int ROWS = BATCH * NTOK;      // 1154
    const int BH   = BATCH * NH;        // 16
    const int ATTR = BH * NTOK;         // 9232

    // ---- workspace carve (all sizes multiples of 4 floats -> 16B aligned) ----
    float* p = (float*)d_ws;
    auto take = [&](size_t n) { float* r = p; p += n; return r; };
    float* x    = take((size_t)ROWS * EMB);
    float* hbuf = take((size_t)ROWS * EMB);
    float* qbuf = take((size_t)ROWS * EMB);
    float* kbuf = take((size_t)ROWS * EMB);
    float* vbuf = take((size_t)ROWS * EMB);
    float* gbuf = take((size_t)ROWS * 3 * NH);
    float* kw   = take((size_t)BH * NCMP * CBK * DHD);
    float* vw   = take((size_t)BH * NCMP * CBK * DHD);
    float* ck   = take((size_t)BH * NCMP * DHD);
    float* cv   = take((size_t)BH * NCMP * DHD);
    float* sc   = take((size_t)BH * NTOK * NCMPP);   // padded prob rows
    int*   selb = (int*)take((size_t)ATTR * KSEL4);
    float* oc   = take((size_t)BH * NTOK * DHD);
    float* mg   = take((size_t)ROWS * EMB);
    float* ffh  = take((size_t)ROWS * MLPD);
    _Float16* bpack = (_Float16*)take(1 << 19);      // 1M f16 pack scratch (reused)

    // pack B then run WMMA GEMM, stream-ordered into shared pack scratch.
    // Kloop: padded K (mult 32, A must be readable/zero there);
    // Kb: true source-K guard for packing.
    auto gemm = [&](const float* A, const float* Bsrc, const float* bias,
                    const float* res, float* C,
                    int M, int N, int Kloop, int Kb,
                    int lda, int ldbK, int ldbN, int ldc,
                    int batches, int innerB,
                    long long sAb, long long sAh, long long sBb, long long sBh,
                    long long sCb, long long sCh, int bBatches,
                    float alpha, int flags) {
        int Kp   = (Kloop + 31) & ~31;
        int Np64 = (N + 63) & ~63;
        long long sBp = (long long)Np64 * Kp;
        int ptot = Np64 * Kp;
        dim3 pg((ptot + 255) / 256, 1, bBatches);
        pack_b<<<pg, 256, 0, stream>>>(Bsrc, bpack, Kb, N, Kp, Np64,
                                       ldbK, ldbN, innerB, sBb, sBh, sBp);
        dim3 gr(Np64 / 64, (M + 15) / 16, batches);
        wmma_gemm<<<gr, dim3(32), 0, stream>>>(A, bpack, bias, res, C,
            M, N, Kp, lda, ldc, innerB,
            sAb, sAh, (bBatches > 1 ? sBp : 0), sCb, sCh, alpha, flags);
    };

    patch_embed<<<ROWS, 256, 0, stream>>>(img, patch_w, patch_b, pos_emb,
                                          cls_tok, x);

    for (int l = 0; l < NDEPTH; ++l) {
        const float* Wq_l  = Wq  + (size_t)l * EMB * EMB;
        const float* Wk_l  = Wk  + (size_t)l * EMB * EMB;
        const float* Wv_l  = Wv  + (size_t)l * EMB * EMB;
        const float* Wg_l  = Wg  + (size_t)l * EMB * 3 * NH;
        const float* Wo_l  = Wo  + (size_t)l * EMB * EMB;
        const float* kpe_l = kpe + (size_t)l * CBK * DHD;
        const float* vpe_l = vpe + (size_t)l * CBK * DHD;
        const float* Wkc_l = Wkc + (size_t)l * CBK * DHD * DHD;
        const float* Wvc_l = Wvc + (size_t)l * CBK * DHD * DHD;
        const float* w1_l  = ff_w1 + (size_t)l * EMB * MLPD;
        const float* b1_l  = ff_b1 + (size_t)l * MLPD;
        const float* w2_l  = ff_w2 + (size_t)l * MLPD * EMB;
        const float* b2_l  = ff_b2 + (size_t)l * EMB;

        ln_kernel<<<ROWS, 256, 0, stream>>>(x, ln1_g + l * EMB, ln1_b + l * EMB,
                                            hbuf, EMB, EMB);

        gemm(hbuf, Wq_l, nullptr, nullptr, qbuf, ROWS, EMB, EMB, EMB,
             EMB, EMB, 1, EMB, 1, 1, 0, 0, 0, 0, 0, 0, 1, 1.f, 0);
        gemm(hbuf, Wk_l, nullptr, nullptr, kbuf, ROWS, EMB, EMB, EMB,
             EMB, EMB, 1, EMB, 1, 1, 0, 0, 0, 0, 0, 0, 1, 1.f, 0);
        gemm(hbuf, Wv_l, nullptr, nullptr, vbuf, ROWS, EMB, EMB, EMB,
             EMB, EMB, 1, EMB, 1, 1, 0, 0, 0, 0, 0, 0, 1, 1.f, 0);
        gemm(hbuf, Wg_l, nullptr, nullptr, gbuf, ROWS, 3 * NH, EMB, EMB,
             EMB, 3 * NH, 1, 3 * NH, 1, 1, 0, 0, 0, 0, 0, 0, 1, 1.f, F_SIG);

        {
            size_t tot = (size_t)BH * NCMP * CBK * DHD;
            build_kw_vw<<<(unsigned)((tot + 255) / 256), 256, 0, stream>>>(
                kbuf, vbuf, kpe_l, vpe_l, kw, vw);
        }

        // ck/cv = kw/vw (145 x 1024) @ Wkc/Wvc (1024 x 64), B shared, batch 16
        gemm(kw, Wkc_l, nullptr, nullptr, ck, NCMP, DHD, CBK * DHD, CBK * DHD,
             CBK * DHD, DHD, 1, DHD, BH, 1,
             (long long)NCMP * CBK * DHD, 0, 0, 0, (long long)NCMP * DHD, 0,
             1, 1.f, 0);
        gemm(vw, Wvc_l, nullptr, nullptr, cv, NCMP, DHD, CBK * DHD, CBK * DHD,
             CBK * DHD, DHD, 1, DHD, BH, 1,
             (long long)NCMP * CBK * DHD, 0, 0, 0, (long long)NCMP * DHD, 0,
             1, 1.f, 0);

        // sc = scale * q (577 x 64) @ ck^T (64 x 145), batched (b,h)
        gemm(qbuf, ck, nullptr, nullptr, sc, NTOK, NCMP, DHD, DHD,
             EMB, /*ldbK=*/1, /*ldbN=*/DHD, NCMPP, BH, NH,
             (long long)NTOK * EMB, DHD,
             (long long)NH * NCMP * DHD, (long long)NCMP * DHD,
             (long long)NH * NTOK * NCMPP, (long long)NTOK * NCMPP,
             BH, 0.125f, 0);

        softmax_rows<<<ATTR, 256, 0, stream>>>(sc, NCMP, NCMPP);
        topk_kernel<<<(ATTR + 255) / 256, 256, 0, stream>>>(sc, selb, ATTR);

        // out_c = pc (577 x 160pad) @ cv (145 x 64), batch 16
        gemm(sc, cv, nullptr, nullptr, oc, NTOK, DHD, NCMPP, NCMP,
             NCMPP, DHD, 1, DHD, BH, 1,
             (long long)NTOK * NCMPP, 0, (long long)NCMP * DHD, 0,
             (long long)NTOK * DHD, 0, BH, 1.f, 0);

        attn_combine<<<ATTR, 256, 0, stream>>>(qbuf, kbuf, vbuf, oc, gbuf,
                                               selb, mg);

        // x = mg @ Wo + x (fused residual, in-place safe)
        gemm(mg, Wo_l, nullptr, x, x, ROWS, EMB, EMB, EMB,
             EMB, EMB, 1, EMB, 1, 1, 0, 0, 0, 0, 0, 0, 1, 1.f, F_RES);

        ln_kernel<<<ROWS, 256, 0, stream>>>(x, ln2_g + l * EMB, ln2_b + l * EMB,
                                            hbuf, EMB, EMB);

        gemm(hbuf, w1_l, b1_l, nullptr, ffh, ROWS, MLPD, EMB, EMB,
             EMB, MLPD, 1, MLPD, 1, 1, 0, 0, 0, 0, 0, 0, 1, 1.f,
             F_BIAS | F_GELU);
        gemm(ffh, w2_l, b2_l, x, x, ROWS, EMB, MLPD, MLPD,
             MLPD, EMB, 1, EMB, 1, 1, 0, 0, 0, 0, 0, 0, 1, 1.f,
             F_BIAS | F_RES);
    }

    // head: LN on cls rows, then (2 x 512) @ (512 x 1000) + bias
    ln_kernel<<<BATCH, 256, 0, stream>>>(x, hln_g, hln_b, hbuf,
                                         (long long)NTOK * EMB, EMB);
    gemm(hbuf, head_w, head_b, nullptr, out, BATCH, NCLS, EMB, EMB,
         EMB, NCLS, 1, NCLS, 1, 1, 0, 0, 0, 0, 0, 0, 1, 1.f, F_BIAS);
}
